// ForwardModel_39462159515831
// MI455X (gfx1250) — matrix operand order repaired and checked by reading, not compiled, not run on hardware
//
#include <hip/hip_runtime.h>

// ---------------------------------------------------------------------------
// SRU forward: L=4, B=32, T=512, D=512, V=1000
//   per layer: U = x @ W[l]  (16384 x 512) x (512 x 1536)   -> WMMA bf16
//              sequential scan over t (elementwise, f32)
// GEMM: double-buffered async global->LDS (ASYNCcnt) pipeline overlapped
// with v_wmma_f32_16x16x32_bf16.
// ---------------------------------------------------------------------------

#define L_ 4
#define B_ 32
#define T_ 512
#define D_ 512
#define V_ 1000
#define N3_ 1536          // 3*D
#define M_ 16384          // T*B
#define NCHUNK 16         // 512 / 32 K-chunks

typedef __attribute__((ext_vector_type(16))) __bf16   v16bf;
typedef __attribute__((ext_vector_type(8)))  float    v8f;
typedef __attribute__((ext_vector_type(4)))  float    f32x4;
typedef __attribute__((ext_vector_type(4)))  __bf16   bf16x4;
typedef __attribute__((ext_vector_type(2)))  unsigned uint2v;
typedef __attribute__((ext_vector_type(4)))  unsigned uint4v;

union FragA { uint2v u2[4]; v16bf v; };   // 32 bytes: 8 VGPRs of packed bf16
union FragB { uint4v u4[2]; v16bf v; };   // 32 bytes

// ---------------------------------------------------------------------------
// W (L,D,3D) f32 -> bf16, vectorized x4
// ---------------------------------------------------------------------------
__global__ __launch_bounds__(256) void sru_wconv(const float* __restrict__ W,
                                                 __bf16* __restrict__ Wh) {
    int g = blockIdx.x * blockDim.x + threadIdx.x;   // over (L*D*3D)/4
    const f32x4 w = *(const f32x4*)(W + (size_t)g * 4);
    bf16x4 o;
    o[0] = (__bf16)w[0]; o[1] = (__bf16)w[1];
    o[2] = (__bf16)w[2]; o[3] = (__bf16)w[3];
    *(bf16x4*)(Wh + (size_t)g * 4) = o;
}

// ---------------------------------------------------------------------------
// embedding gather: x[(t*B+b)*D + d] = emb[ids[b,t]*D + d]  (f32 + bf16 copy)
// ---------------------------------------------------------------------------
__global__ __launch_bounds__(256) void sru_embed(const int* __restrict__ ids,
                                                 const float* __restrict__ emb,
                                                 float* __restrict__ x,
                                                 __bf16* __restrict__ xh) {
    int g  = blockIdx.x * blockDim.x + threadIdx.x;  // over T*B*D/4
    int d4 = (g % (D_ / 4)) * 4;
    int tb = g / (D_ / 4);                           // t*B + b
    int t  = tb / B_;
    int b  = tb % B_;
    int id = ids[b * T_ + t];
    const f32x4 e = *(const f32x4*)(emb + (size_t)id * D_ + d4);
    *(f32x4*)(x + (size_t)tb * D_ + d4) = e;
    bf16x4 h;
    h[0] = (__bf16)e[0]; h[1] = (__bf16)e[1];
    h[2] = (__bf16)e[2]; h[3] = (__bf16)e[3];
    *(bf16x4*)(xh + (size_t)tb * D_ + d4) = h;
}

// ---------------------------------------------------------------------------
// GEMM: U(M x N3) = A(M x 512, bf16) * Bm(512 x N3, bf16), f32 accumulate.
// Block tile 256(M) x 64(N), K-chunk 32, double-buffered LDS. 8 waves; wave w
// owns rows [w*32, w*32+32) x 64 cols => 2x4 wmma tiles (B-frag reused 2x).
// Pipeline: wait(own async) -> barrier(publish) -> issue next chunk -> wmma.
// ---------------------------------------------------------------------------
__global__ __launch_bounds__(256) void sru_gemm(const __bf16* __restrict__ A,
                                                const __bf16* __restrict__ Bm,
                                                float* __restrict__ U) {
    __shared__ __bf16 As[2][256][40];   // 32 + 8 pad (80B row, 16B aligned), 20480B/buf
    __shared__ __bf16 Bs[2][32][72];    // 64 + 8 pad (144B row, 16B aligned), 4608B/buf

    const int tid  = threadIdx.x;
    const int lane = tid & 31;
    const int wave = tid >> 5;
    const int gm0  = blockIdx.x * 256;
    const int gn0  = blockIdx.y * 64;

    // cooperative-load coordinates (256 threads, 4 A rows + 1 B row each)
    const int ar = tid >> 2;          // 0..63   (A rows +0,+64,+128,+192)
    const int ac = (tid & 3) << 3;    // 0,8,16,24
    const int br = tid >> 3;          // 0..31   (B rows)
    const int bc = (tid & 7) << 3;    // 0..56

    // async-load LDS destinations for buffer 0 (byte offsets); buffer 1 = +size
    const unsigned ldsA0 = (unsigned)(size_t)&As[0][ar][ac];
    const unsigned ldsA1 = (unsigned)(size_t)&As[0][ar + 64][ac];
    const unsigned ldsA2 = (unsigned)(size_t)&As[0][ar + 128][ac];
    const unsigned ldsA3 = (unsigned)(size_t)&As[0][ar + 192][ac];
    const unsigned ldsB  = (unsigned)(size_t)&Bs[0][br][bc];
    const unsigned ABUF  = 256u * 40u * 2u;   // 20480
    const unsigned BBUF  = 32u * 72u * 2u;    //  4608

    const unsigned long long gAbase =
        (unsigned long long)(A + (size_t)(gm0 + ar) * D_ + ac);
    const unsigned long long gBbase =
        (unsigned long long)(Bm + (size_t)br * N3_ + gn0 + bc);

    // issue one K-chunk's staging loads into buffer p (5 async b128 per thread)
    auto issue = [&](int kc, int p) {
        const unsigned long long gA = gAbase + (unsigned long long)kc * 2;
        const unsigned long long gB = gBbase + (unsigned long long)kc * (N3_ * 2);
        const unsigned pa = (unsigned)p * ABUF;
        const unsigned pb = (unsigned)p * BBUF;
        asm volatile("global_load_async_to_lds_b128 %0, %1, off"
                     :: "v"(ldsA0 + pa), "v"(gA) : "memory");
        asm volatile("global_load_async_to_lds_b128 %0, %1, off"
                     :: "v"(ldsA1 + pa), "v"(gA + 64ull * D_ * 2) : "memory");
        asm volatile("global_load_async_to_lds_b128 %0, %1, off"
                     :: "v"(ldsA2 + pa), "v"(gA + 128ull * D_ * 2) : "memory");
        asm volatile("global_load_async_to_lds_b128 %0, %1, off"
                     :: "v"(ldsA3 + pa), "v"(gA + 192ull * D_ * 2) : "memory");
        asm volatile("global_load_async_to_lds_b128 %0, %1, off"
                     :: "v"(ldsB + pb), "v"(gB) : "memory");
    };

    // fragment coordinates (ISA 16-bit A layout: lanes 0-15 K-low, 16-31 K-high)
    const int lrow = lane & 15;
    const int khi  = (lane >> 4) << 3;   // 0 or 8
    const __bf16* Abase0 = &As[0][wave * 32 + lrow][0];
    const __bf16* Bbase0 = &Bs[0][0][lane & 15];

    v8f acc[2][4] = {};

    issue(0, 0);                          // prologue: chunk 0 -> buffer 0

    for (int i = 0; i < NCHUNK; ++i) {
        const int p = i & 1;
        asm volatile("s_wait_asynccnt 0" ::: "memory");  // own chunk-i loads landed
        __syncthreads();                  // publish buf p; buf p^1 free everywhere
        if (i + 1 < NCHUNK)
            issue((i + 1) * 32, p ^ 1);   // overlap next DMA with this compute
        if (i + 2 < NCHUNK) {             // keep L2/near caches ahead of the DMA
            __builtin_prefetch(A + (size_t)(gm0 + ar) * D_ + (i + 2) * 32 + ac, 0, 3);
            __builtin_prefetch(Bm + (size_t)((i + 2) * 32 + br) * N3_ + gn0 + bc, 0, 3);
        }

        // two A fragments (rows wave*32 .. +15, +16 .. +31)
        // VGPR pairs (0,1)=K 0..3|8..11, (2,3)=K 4..7|12..15,
        //            (4,5)=K 16..19|24..27, (6,7)=K 20..23|28..31
        const __bf16* ar0 = Abase0 + p * (256 * 40);
        const __bf16* ar1 = ar0 + 16 * 40;
        FragA a0, a1;
        a0.u2[0] = *(const uint2v*)(ar0 + khi);
        a0.u2[1] = *(const uint2v*)(ar0 + khi + 4);
        a0.u2[2] = *(const uint2v*)(ar0 + 16 + khi);
        a0.u2[3] = *(const uint2v*)(ar0 + 16 + khi + 4);
        a1.u2[0] = *(const uint2v*)(ar1 + khi);
        a1.u2[1] = *(const uint2v*)(ar1 + khi + 4);
        a1.u2[2] = *(const uint2v*)(ar1 + 16 + khi);
        a1.u2[3] = *(const uint2v*)(ar1 + 16 + khi + 4);

        const __bf16* brow0 = Bbase0 + p * (32 * 72);
#pragma unroll
        for (int nt = 0; nt < 4; ++nt) {
            // B fragment: lane%16 = N col, element s = K row (same K order as A)
            FragB bf_;
#pragma unroll
            for (int s = 0; s < 16; ++s) {
                const int kk = (s < 8) ? (khi + s) : (16 + khi + (s - 8));
                bf_.v[s] = brow0[kk * 72 + nt * 16];
            }
            acc[0][nt] = __builtin_amdgcn_wmma_f32_16x16x32_bf16(
                false, a0.v, false, bf_.v, (short)0, acc[0][nt], false, false);
            acc[1][nt] = __builtin_amdgcn_wmma_f32_16x16x32_bf16(
                false, a1.v, false, bf_.v, (short)0, acc[1][nt], false, false);
        }
    }

    // store: C/D layout: VGPR r -> M = r (lanes 0-15) / r+8 (lanes 16-31), N = lane%16
    const int n  = lane & 15;
    const int mb = (lane < 16) ? 0 : 8;
#pragma unroll
    for (int mi = 0; mi < 2; ++mi) {
#pragma unroll
        for (int nt = 0; nt < 4; ++nt) {
#pragma unroll
            for (int r = 0; r < 8; ++r) {
                U[(size_t)(gm0 + wave * 32 + mi * 16 + mb + r) * N3_ +
                  gn0 + nt * 16 + n] = acc[mi][nt][r];
            }
        }
    }
}

// ---------------------------------------------------------------------------
// SRU recurrence. One thread per (b,d); loops t = 0..T-1. Updates x/xh
// in place (x[t] is dead after h[t] is produced for this (b,d)).
// Last layer also writes d_out in (B,T,D) order.
// ---------------------------------------------------------------------------
__global__ __launch_bounds__(256) void sru_scan(const float* __restrict__ U,
                                                float* __restrict__ x,
                                                __bf16* __restrict__ xh,
                                                const unsigned char* __restrict__ mask,
                                                const float* __restrict__ vf,
                                                const float* __restrict__ vr,
                                                const float* __restrict__ bfp,
                                                const float* __restrict__ brp,
                                                float* __restrict__ out,
                                                int last) {
    int g = blockIdx.x * blockDim.x + threadIdx.x;   // 0 .. B*D-1
    int b = g / D_;
    int d = g % D_;
    const float vfd = vf[d], vrd = vr[d], bfd = bfp[d], brd = brp[d];
    const unsigned char* mrow = mask + (size_t)b * T_;
    float c = 0.0f;
    for (int t = 0; t < T_; ++t) {
        const size_t row = (size_t)(t * B_ + b);
        const float* u = U + row * N3_;
        const float xt = u[d];
        const float fp = u[D_ + d];
        const float rp = u[2 * D_ + d];
        const float xv = x[row * D_ + d];
        const float f  = 1.0f / (1.0f + __expf(-(fp + vfd * c + bfd)));
        const float r  = 1.0f / (1.0f + __expf(-(rp + vrd * c + brd)));
        const float cn = f * c + (1.0f - f) * xt;
        float h        = r * tanhf(cn) + (1.0f - r) * xv;
        const bool pad = (mrow[t] == 0);
        c = pad ? c : cn;
        h = pad ? 0.0f : h;
        x[row * D_ + d]  = h;
        xh[row * D_ + d] = (__bf16)h;
        if (last) out[((size_t)b * T_ + t) * D_ + d] = h;
    }
}

// ---------------------------------------------------------------------------
// launch
// ---------------------------------------------------------------------------
extern "C" void kernel_launch(void* const* d_in, const int* in_sizes, int n_in,
                              void* d_out, int out_size, void* d_ws, size_t ws_size,
                              hipStream_t stream) {
    const int*           ids  = (const int*)d_in[0];            // (B,T) int32
    const unsigned char* mask = (const unsigned char*)d_in[1];  // (B,T) bool
    const float*         emb  = (const float*)d_in[2];          // (V,D)
    const float*         W    = (const float*)d_in[3];          // (L,D,3D)
    const float*         v    = (const float*)d_in[4];          // (L,2,D)
    const float*         bb   = (const float*)d_in[5];          // (L,2,D)
    float* out = (float*)d_out;                                 // (B,T,D)

    char* ws = (char*)d_ws;
    const size_t WH_BYTES = (size_t)L_ * D_ * N3_ * 2;          //   6,291,456
    const size_t X_BYTES  = (size_t)M_ * D_ * 4;                //  33,554,432
    const size_t XH_BYTES = (size_t)M_ * D_ * 2;                //  16,777,216
    __bf16* Wh = (__bf16*)ws;
    float*  x  = (float*)(ws + WH_BYTES);
    __bf16* xh = (__bf16*)(ws + WH_BYTES + X_BYTES);
    float*  U  = (float*)(ws + WH_BYTES + X_BYTES + XH_BYTES);  // 100,663,296

    // W -> bf16 : (L*D*3D)/4 / 256 = 3072 blocks
    sru_wconv<<<3072, 256, 0, stream>>>(W, Wh);
    // embedding gather: (T*B*D)/4 / 256 = 8192 blocks
    sru_embed<<<8192, 256, 0, stream>>>(ids, emb, x, xh);

    for (int l = 0; l < L_; ++l) {
        sru_gemm<<<dim3(M_ / 256, N3_ / 64), 256, 0, stream>>>(
            xh, Wh + (size_t)l * D_ * N3_, U);
        sru_scan<<<(B_ * D_) / 256, 256, 0, stream>>>(
            U, x, xh, mask,
            v  + (size_t)(l * 2 + 0) * D_, v  + (size_t)(l * 2 + 1) * D_,
            bb + (size_t)(l * 2 + 0) * D_, bb + (size_t)(l * 2 + 1) * D_,
            out, (l == L_ - 1) ? 1 : 0);
    }
}
